// SelfAttention_22668837388400
// MI455X (gfx1250) — matrix operand order repaired and checked by reading, not compiled, not run on hardware
//
#include <hip/hip_runtime.h>
#include <hip/hip_bf16.h>

// ---------------- problem constants ----------------
constexpr int Bc = 4, Nc = 2048, Dc = 1024, Ec = 64, Hc = 16;
constexpr int HEc = Hc * Ec; // 1024

// ---------------- types ----------------
typedef __attribute__((ext_vector_type(16))) __bf16 v16bf;
typedef __attribute__((ext_vector_type(8)))  float  v8f;
typedef __attribute__((ext_vector_type(4)))  int    i4v;

union V16U { v16bf v; i4v q[2]; };

#define DEVINL static __device__ __forceinline__

DEVINL unsigned short f2bf(float f) {
  unsigned int u = __float_as_uint(f);
  u += 0x7FFFu + ((u >> 16) & 1u);   // round-to-nearest-even
  return (unsigned short)(u >> 16);
}

// A operand: 16(M) x 32(K) bf16 tile, source row-major with leading dim ld.
// lane<16 : row=lane,    K = {k0..k0+7, k0+16..k0+23}
// lane>=16: row=lane-16, K = {k0+8..k0+15, k0+24..k0+31}
DEVINL v16bf load_a16x32(const unsigned short* p0, int ld) {
  int lane = threadIdx.x & 31;
  const unsigned short* p = p0 + (size_t)(lane & 15) * ld + ((lane >> 4) << 3);
  V16U u;
  u.q[0] = *(const i4v*)(p);
  u.q[1] = *(const i4v*)(p + 16);
  return u.v;
}

// B operand: 32(K) x 16(N) bf16 tile. Source is B^T row-major (row = column n),
// leading dim ld. lane<16: col=lane, K0..7,16..23 ; lane>=16: col=lane-16, K8..15,24..31.
DEVINL v16bf load_b32x16(const unsigned short* p0, int ld) {
  int lane = threadIdx.x & 31;
  const unsigned short* p = p0 + (size_t)(lane & 15) * ld + ((lane >> 4) << 3);
  V16U u;
  u.q[0] = *(const i4v*)(p);
  u.q[1] = *(const i4v*)(p + 16);
  return u.v;
}

DEVINL v8f wmma_bf16(v16bf a, v16bf b, v8f c) {
  return __builtin_amdgcn_wmma_f32_16x16x32_bf16(false, a, false, b, (short)0, c, false, false);
}

// ---------------- pack kernels ----------------
__global__ void k_pack_x(const float* __restrict__ x, unsigned short* __restrict__ xb) {
  size_t i = (size_t)blockIdx.x * blockDim.x + threadIdx.x;
  xb[i] = f2bf(x[i]);
}

// wt[h][e][d] = w[h][d][e]   (over H*E*D)
__global__ void k_pack_wT(const float* __restrict__ w, unsigned short* __restrict__ wt) {
  size_t i = (size_t)blockIdx.x * blockDim.x + threadIdx.x;
  int d = (int)(i % Dc);
  int e = (int)((i / Dc) % Ec);
  int h = (int)(i / ((size_t)Dc * Ec));
  wt[i] = f2bf(w[((size_t)h * Dc + d) * Ec + e]);
}

// wat[d][he] = w[he][d]      (over D*HE)
__global__ void k_pack_aggT(const float* __restrict__ w, unsigned short* __restrict__ wt) {
  size_t i = (size_t)blockIdx.x * blockDim.x + threadIdx.x;
  int he = (int)(i % HEc);
  int d  = (int)(i / HEc);
  wt[i] = f2bf(w[(size_t)he * Dc + d]);
}

// ---------------- projections: 32(n) x 64(e) per wave, ping-pong double buffered ----------------
// Q,K row-major [bh][n][e] (Q scaled by 1/sqrt(E)); V stored transposed [bh][e][n]
__global__ void __launch_bounds__(128)
k_proj(const unsigned short* __restrict__ xb,
       const unsigned short* __restrict__ wqt, const unsigned short* __restrict__ wkt,
       const unsigned short* __restrict__ wvt,
       unsigned short* __restrict__ qb, unsigned short* __restrict__ kb,
       unsigned short* __restrict__ vtb) {
  const int which = blockIdx.z;              // 0=Q 1=K 2=V
  const int bh = blockIdx.y;                 // b*H+h
  const int b = bh >> 4, h = bh & 15;
  const int wave = threadIdx.x >> 5, lane = threadIdx.x & 31;
  const int n0 = (blockIdx.x * 4 + wave) * 32;

  const unsigned short* w = (which == 0) ? wqt : (which == 1) ? wkt : wvt;
  const unsigned short* wt_h = w + (size_t)h * Ec * Dc;     // [e][d]
  const unsigned short* xrow = xb + ((size_t)b * Nc + n0) * Dc;

  v8f acc[2][4] = {};
  v16bf xa0, xa1, xbm[4];                    // buffer X
  v16bf ya0, ya1, ybm[4];                    // buffer Y
  // prologue: fill X with dc=0
  xa0 = load_a16x32(xrow, Dc);
  xa1 = load_a16x32(xrow + (size_t)16 * Dc, Dc);
#pragma unroll
  for (int eb = 0; eb < 4; ++eb) xbm[eb] = load_b32x16(wt_h + (size_t)(eb * 16) * Dc, Dc);

  for (int dc = 0; dc < Dc; dc += 64) {
    const int p1 = dc + 32;                  // always < Dc
    ya0 = load_a16x32(xrow + p1, Dc);
    ya1 = load_a16x32(xrow + (size_t)16 * Dc + p1, Dc);
#pragma unroll
    for (int eb = 0; eb < 4; ++eb) ybm[eb] = load_b32x16(wt_h + (size_t)(eb * 16) * Dc + p1, Dc);
#pragma unroll
    for (int eb = 0; eb < 4; ++eb) {
      acc[0][eb] = wmma_bf16(xa0, xbm[eb], acc[0][eb]);
      acc[1][eb] = wmma_bf16(xa1, xbm[eb], acc[1][eb]);
    }
    const int p2 = (dc + 64) & (Dc - 1);     // dead wrap on last iter
    xa0 = load_a16x32(xrow + p2, Dc);
    xa1 = load_a16x32(xrow + (size_t)16 * Dc + p2, Dc);
#pragma unroll
    for (int eb = 0; eb < 4; ++eb) xbm[eb] = load_b32x16(wt_h + (size_t)(eb * 16) * Dc + p2, Dc);
#pragma unroll
    for (int eb = 0; eb < 4; ++eb) {
      acc[0][eb] = wmma_bf16(ya0, ybm[eb], acc[0][eb]);
      acc[1][eb] = wmma_bf16(ya1, ybm[eb], acc[1][eb]);
    }
  }

  const int col = lane & 15;
  const int rbase = (lane >> 4) << 3;        // D-layout: row = r + 8*(lane/16)
  const float scale = (which == 0) ? 0.125f : 1.0f;  // fold 1/sqrt(E) into Q
  if (which < 2) {
    unsigned short* dst = ((which == 0) ? qb : kb) + ((size_t)bh * Nc + n0) * Ec;
#pragma unroll
    for (int t = 0; t < 2; ++t)
#pragma unroll
      for (int eb = 0; eb < 4; ++eb)
#pragma unroll
        for (int r = 0; r < 8; ++r)
          dst[(size_t)(t * 16 + rbase + r) * Ec + eb * 16 + col] = f2bf(acc[t][eb][r] * scale);
  } else {
    unsigned short* dst = vtb + (size_t)bh * Ec * Nc;       // [e][n]
#pragma unroll
    for (int t = 0; t < 2; ++t)
#pragma unroll
      for (int eb = 0; eb < 4; ++eb)
#pragma unroll
        for (int r = 0; r < 8; ++r)
          dst[(size_t)(eb * 16 + col) * Nc + (n0 + t * 16 + rbase + r)] = f2bf(acc[t][eb][r]);
  }
}

// ---------------- pass 1: per-key-column softmax stats (softmax over q!) ----------------
// 32 key columns per wave; Q-tiles ping-pong double buffered; K fragments loop-invariant.
__global__ void __launch_bounds__(128)
k_stats(const unsigned short* __restrict__ qb, const unsigned short* __restrict__ kb,
        float* __restrict__ mout, float* __restrict__ rzout) {
  const int bh = blockIdx.y;
  const int wave = threadIdx.x >> 5, lane = threadIdx.x & 31;
  const int k0 = (blockIdx.x * 4 + wave) * 32;             // 32 key columns per wave
  const unsigned short* qbase = qb + (size_t)bh * Nc * Ec;
  const unsigned short* kbase = kb + (size_t)bh * Nc * Ec;

  v16bf bkf[4];                                            // [j][echunk], loop invariant
  bkf[0] = load_b32x16(kbase + (size_t)k0 * Ec + 0, Ec);
  bkf[1] = load_b32x16(kbase + (size_t)k0 * Ec + 32, Ec);
  bkf[2] = load_b32x16(kbase + (size_t)(k0 + 16) * Ec + 0, Ec);
  bkf[3] = load_b32x16(kbase + (size_t)(k0 + 16) * Ec + 32, Ec);

  float m_run[2] = {-__builtin_inff(), -__builtin_inff()};
  float s_run[2] = {0.f, 0.f};

  auto update = [&](v16bf a0, v16bf a1) {
#pragma unroll
    for (int j = 0; j < 2; ++j) {
      v8f s = {};
      s = wmma_bf16(a0, bkf[2 * j + 0], s);
      s = wmma_bf16(a1, bkf[2 * j + 1], s);
      float tmax = s[0];
#pragma unroll
      for (int r = 1; r < 8; ++r) tmax = fmaxf(tmax, s[r]);
      float nm = fmaxf(m_run[j], tmax);
      float ssum = 0.f;
#pragma unroll
      for (int r = 0; r < 8; ++r) ssum += __expf(s[r] - nm);
      s_run[j] = s_run[j] * __expf(m_run[j] - nm) + ssum;
      m_run[j] = nm;
    }
  };

  v16bf xa0 = load_a16x32(qbase + 0, Ec);
  v16bf xa1 = load_a16x32(qbase + 32, Ec);
  v16bf ya0, ya1;
  for (int q0 = 0; q0 < Nc; q0 += 32) {
    const int p1 = q0 + 16;                                // always < Nc
    ya0 = load_a16x32(qbase + (size_t)p1 * Ec + 0, Ec);
    ya1 = load_a16x32(qbase + (size_t)p1 * Ec + 32, Ec);
    update(xa0, xa1);
    const int p2 = (q0 + 32) & (Nc - 1);                   // dead wrap on last iter
    xa0 = load_a16x32(qbase + (size_t)p2 * Ec + 0, Ec);
    xa1 = load_a16x32(qbase + (size_t)p2 * Ec + 32, Ec);
    update(ya0, ya1);
  }
  // combine lane halves (rows 0-7 handled by lane, rows 8-15 by lane^16; same column)
#pragma unroll
  for (int j = 0; j < 2; ++j) {
    float m2 = __shfl_xor(m_run[j], 16, 32);
    float s2 = __shfl_xor(s_run[j], 16, 32);
    float nm = fmaxf(m_run[j], m2);
    float st = s_run[j] * __expf(m_run[j] - nm) + s2 * __expf(m2 - nm);
    if (lane < 16) {
      mout[(size_t)bh * Nc + k0 + j * 16 + lane] = nm;
      rzout[(size_t)bh * Nc + k0 + j * 16 + lane] = 1.0f / st;
    }
  }
}

// ---------------- pass 2: heads = (exp(S - m_k)/Z_k) @ V, streamed over k ----------------
// 32 query rows x 64 e per wave; all global loads of a k-chunk hoisted to iteration start.
__global__ void __launch_bounds__(128)
k_attn(const unsigned short* __restrict__ qb, const unsigned short* __restrict__ kb,
       const unsigned short* __restrict__ vtb,
       const float* __restrict__ mcol, const float* __restrict__ rzcol,
       unsigned short* __restrict__ hc) {
  __shared__ unsigned short sP[4][2][16 * 32];             // per-wave P tiles (D->A layout bounce)
  const int bh = blockIdx.y;
  const int b = bh >> 4, h = bh & 15;
  const int wave = threadIdx.x >> 5, lane = threadIdx.x & 31;
  const int q0 = (blockIdx.x * 4 + wave) * 32;

  const unsigned short* qbase = qb + (size_t)bh * Nc * Ec;
  const unsigned short* kbase = kb + (size_t)bh * Nc * Ec;
  const unsigned short* vbase = vtb + (size_t)bh * Ec * Nc;
  const float* mb = mcol + (size_t)bh * Nc;
  const float* rb = rzcol + (size_t)bh * Nc;

  v16bf aq[2][2];                                          // [qsub][echunk], loop invariant
#pragma unroll
  for (int t = 0; t < 2; ++t) {
    aq[t][0] = load_a16x32(qbase + (size_t)(q0 + t * 16) * Ec + 0, Ec);
    aq[t][1] = load_a16x32(qbase + (size_t)(q0 + t * 16) * Ec + 32, Ec);
  }

  v8f hacc[2][4] = {};
  unsigned short* sp0 = sP[wave][0];
  unsigned short* sp1 = sP[wave][1];
  const int col = lane & 15;
  const int rbase = (lane >> 4) << 3;

  for (int kc = 0; kc < Nc; kc += 32) {
    // issue ALL global loads for this chunk up front (in flight during scores/exp/LDS)
    v16bf bv[4];
#pragma unroll
    for (int eb = 0; eb < 4; ++eb) bv[eb] = load_b32x16(vbase + (size_t)(eb * 16) * Nc + kc, Nc);
    v16bf bk[2][2];
#pragma unroll
    for (int kblk = 0; kblk < 2; ++kblk) {
      bk[kblk][0] = load_b32x16(kbase + (size_t)(kc + kblk * 16) * Ec + 0, Ec);
      bk[kblk][1] = load_b32x16(kbase + (size_t)(kc + kblk * 16) * Ec + 32, Ec);
    }
    float mc[2], rc[2];
    mc[0] = mb[kc + col];       rc[0] = rb[kc + col];
    mc[1] = mb[kc + 16 + col];  rc[1] = rb[kc + 16 + col];

#pragma unroll
    for (int kblk = 0; kblk < 2; ++kblk)
#pragma unroll
      for (int t = 0; t < 2; ++t) {
        v8f s = {};
        s = wmma_bf16(aq[t][0], bk[kblk][0], s);
        s = wmma_bf16(aq[t][1], bk[kblk][1], s);
        unsigned short* sp = t ? sp1 : sp0;
#pragma unroll
        for (int r = 0; r < 8; ++r) {
          float p = __expf(s[r] - mc[kblk]) * rc[kblk];
          sp[(size_t)(rbase + r) * 32 + kblk * 16 + col] = f2bf(p); // [q][k] row-major
        }
      }
    v16bf ap0 = load_a16x32(sp0, 32);                      // P tiles back in A layout (LDS)
    v16bf ap1 = load_a16x32(sp1, 32);
#pragma unroll
    for (int eb = 0; eb < 4; ++eb) {
      hacc[0][eb] = wmma_bf16(ap0, bv[eb], hacc[0][eb]);
      hacc[1][eb] = wmma_bf16(ap1, bv[eb], hacc[1][eb]);
    }
  }

  unsigned short* dst = hc + ((size_t)b * Nc + q0) * HEc + h * Ec; // head-order concat
#pragma unroll
  for (int t = 0; t < 2; ++t)
#pragma unroll
    for (int eb = 0; eb < 4; ++eb)
#pragma unroll
      for (int r = 0; r < 8; ++r)
        dst[(size_t)(t * 16 + rbase + r) * HEc + eb * 16 + col] = f2bf(hacc[t][eb][r]);
}

// ---------------- output aggregation: [B*N,1024] @ [1024,1024] -> fp32, ping-pong ----------------
__global__ void __launch_bounds__(128)
k_agg(const unsigned short* __restrict__ hc, const unsigned short* __restrict__ wat,
      float* __restrict__ out) {
  const int wave = threadIdx.x >> 5, lane = threadIdx.x & 31;
  const int r0 = (blockIdx.x * 4 + wave) * 32;             // row in [0, B*N)
  const int d0 = blockIdx.y * 64;

  v8f acc[2][4] = {};
  v16bf xa0, xa1, xbm[4];
  v16bf ya0, ya1, ybm[4];
  xa0 = load_a16x32(hc + (size_t)r0 * HEc, HEc);
  xa1 = load_a16x32(hc + (size_t)(r0 + 16) * HEc, HEc);
#pragma unroll
  for (int db = 0; db < 4; ++db) xbm[db] = load_b32x16(wat + (size_t)(d0 + db * 16) * HEc, HEc);

  for (int c = 0; c < HEc; c += 64) {
    const int p1 = c + 32;
    ya0 = load_a16x32(hc + (size_t)r0 * HEc + p1, HEc);
    ya1 = load_a16x32(hc + (size_t)(r0 + 16) * HEc + p1, HEc);
#pragma unroll
    for (int db = 0; db < 4; ++db) ybm[db] = load_b32x16(wat + (size_t)(d0 + db * 16) * HEc + p1, HEc);
#pragma unroll
    for (int db = 0; db < 4; ++db) {
      acc[0][db] = wmma_bf16(xa0, xbm[db], acc[0][db]);
      acc[1][db] = wmma_bf16(xa1, xbm[db], acc[1][db]);
    }
    const int p2 = (c + 64) & (HEc - 1);
    xa0 = load_a16x32(hc + (size_t)r0 * HEc + p2, HEc);
    xa1 = load_a16x32(hc + (size_t)(r0 + 16) * HEc + p2, HEc);
#pragma unroll
    for (int db = 0; db < 4; ++db) xbm[db] = load_b32x16(wat + (size_t)(d0 + db * 16) * HEc + p2, HEc);
#pragma unroll
    for (int db = 0; db < 4; ++db) {
      acc[0][db] = wmma_bf16(ya0, ybm[db], acc[0][db]);
      acc[1][db] = wmma_bf16(ya1, ybm[db], acc[1][db]);
    }
  }

  const int col = lane & 15;
  const int rbase = (lane >> 4) << 3;
  float* dst = out + (size_t)r0 * Dc + d0;
#pragma unroll
  for (int t = 0; t < 2; ++t)
#pragma unroll
    for (int db = 0; db < 4; ++db)
#pragma unroll
      for (int r = 0; r < 8; ++r)
        dst[(size_t)(t * 16 + rbase + r) * Dc + db * 16 + col] = acc[t][db][r];
}

// ---------------- workspace layout (bytes) ----------------
constexpr size_t SZ_XB  = (size_t)Bc * Nc * Dc * 2;        // x bf16
constexpr size_t SZ_WT  = (size_t)Hc * Ec * Dc * 2;        // one transposed proj weight
constexpr size_t SZ_WAT = (size_t)Dc * HEc * 2;            // transposed agg weight
constexpr size_t SZ_QKV = (size_t)Bc * Hc * Nc * Ec * 2;   // Q, K, or Vt bf16
constexpr size_t SZ_ST  = (size_t)Bc * Hc * Nc * 4;        // column stats fp32
constexpr size_t SZ_HC  = (size_t)Bc * Nc * HEc * 2;       // concat heads bf16

constexpr size_t O_XB  = 0;
constexpr size_t O_WQT = O_XB  + SZ_XB;
constexpr size_t O_WKT = O_WQT + SZ_WT;
constexpr size_t O_WVT = O_WKT + SZ_WT;
constexpr size_t O_WAT = O_WVT + SZ_WT;
constexpr size_t O_QB  = O_WAT + SZ_WAT;
constexpr size_t O_KB  = O_QB  + SZ_QKV;
constexpr size_t O_VTB = O_KB  + SZ_QKV;
constexpr size_t O_M   = O_VTB + SZ_QKV;
constexpr size_t O_RZ  = O_M   + SZ_ST;
constexpr size_t O_HC  = O_RZ  + SZ_ST;

extern "C" void kernel_launch(void* const* d_in, const int* in_sizes, int n_in,
                              void* d_out, int out_size, void* d_ws, size_t ws_size,
                              hipStream_t stream) {
  (void)in_sizes; (void)n_in; (void)out_size; (void)ws_size;
  const float* x  = (const float*)d_in[0];
  // d_in[1] = attention_mask (unused by the reference math)
  const float* wq = (const float*)d_in[2];
  const float* wk = (const float*)d_in[3];
  const float* wv = (const float*)d_in[4];
  const float* wa = (const float*)d_in[5];
  float* out = (float*)d_out;

  char* ws = (char*)d_ws;
  unsigned short* xb  = (unsigned short*)(ws + O_XB);
  unsigned short* wqt = (unsigned short*)(ws + O_WQT);
  unsigned short* wkt = (unsigned short*)(ws + O_WKT);
  unsigned short* wvt = (unsigned short*)(ws + O_WVT);
  unsigned short* wat = (unsigned short*)(ws + O_WAT);
  unsigned short* qb  = (unsigned short*)(ws + O_QB);
  unsigned short* kb  = (unsigned short*)(ws + O_KB);
  unsigned short* vtb = (unsigned short*)(ws + O_VTB);
  float*          mcol  = (float*)(ws + O_M);
  float*          rzcol = (float*)(ws + O_RZ);
  unsigned short* hc  = (unsigned short*)(ws + O_HC);

  // pack to bf16 (+ transposes for WMMA B-operands)
  k_pack_x   <<<(Bc * Nc * Dc) / 256, 256, 0, stream>>>(x, xb);
  k_pack_wT  <<<(Hc * Ec * Dc) / 256, 256, 0, stream>>>(wq, wqt);
  k_pack_wT  <<<(Hc * Ec * Dc) / 256, 256, 0, stream>>>(wk, wkt);
  k_pack_wT  <<<(Hc * Ec * Dc) / 256, 256, 0, stream>>>(wv, wvt);
  k_pack_aggT<<<(Dc * HEc)     / 256, 256, 0, stream>>>(wa, wat);

  // projections (Q scaled by 1/sqrt(E)), V stored transposed; 32x64 per wave
  k_proj <<<dim3(Nc / 128, Bc * Hc, 3), 128, 0, stream>>>(xb, wqt, wkt, wvt, qb, kb, vtb);

  // pass 1: per-key-column online softmax stats (softmax over query axis); 32 cols/wave
  k_stats<<<dim3(Nc / 128, Bc * Hc), 128, 0, stream>>>(qb, kb, mcol, rzcol);

  // pass 2: streamed attention, no NxN materialization; 32 q-rows/wave
  k_attn <<<dim3(Nc / 128, Bc * Hc), 128, 0, stream>>>(qb, kb, vtb, mcol, rzcol, hc);

  // output aggregation GEMM; 32x64 per wave
  k_agg  <<<dim3((Bc * Nc) / 128, Dc / 64), 128, 0, stream>>>(hc, wat, out);
}
